// SpatialSelfAttention_38663295599080
// MI455X (gfx1250) — compile-verified, matrix operand order checked by reading, not run on hardware
//
#include <hip/hip_runtime.h>

// SpatialSelfAttention for MI455X (gfx1250), compile-only tuned.
// B=4, C=256, H=W=64 -> N=4096, DQK=32.

#define BATCH 4
#define CCH   256
#define NPIX  4096
#define DQKD  32

typedef _Float16 halfx8  __attribute__((ext_vector_type(8)));
typedef _Float16 halfx16 __attribute__((ext_vector_type(16)));
typedef float    floatx8 __attribute__((ext_vector_type(8)));

// ---------------------------------------------------------------------------
// Load a 16x32 f16 tile (row-major, leading dim `ld` in elements) into the
// CDNA5 WMMA 16-bit A-fragment layout. Per CDNA5 ISA:
//   lane L: row = L%16, half = L/16
//   VGPR v holds K pair k = (v/4)*16 + half*8 + (v%4)*2
// => per lane, two contiguous 8-half (16B) runs at k = half*8 and 16+half*8.
// The same layout serves the B operand when fed the tile of B^T (rows = N-dim,
// cols = K-dim), by the A/B layout symmetry.
// ---------------------------------------------------------------------------
__device__ __forceinline__ halfx16 load_tile16x32(const _Float16* base, int ld) {
  const int lane = threadIdx.x & 31;
  const int row  = lane & 15;
  const int hsel = lane >> 4;
  const _Float16* p = base + (size_t)row * ld + hsel * 8;
  halfx8 lo = *(const halfx8*)(p);
  halfx8 hi = *(const halfx8*)(p + 16);
  halfx16 f;
#pragma unroll
  for (int i = 0; i < 8; ++i) { f[i] = lo[i]; f[i + 8] = hi[i]; }
  return f;
}

__device__ __forceinline__ floatx8 wmma_fp16(halfx16 a, halfx16 b, floatx8 c) {
  // (neg_a, A, neg_b, B, c_mod, C, reuse_a, reuse_b)
  return __builtin_amdgcn_wmma_f32_16x16x32_f16(false, a, false, b, (short)0, c,
                                                false, false);
}

// ---------------------------------------------------------------------------
// Stage 1a: x[B,C,N] f32 -> xt[B,N,C] f16, LDS-tiled 16x16 transpose so both
// the global read and the global write are coalesced.
// ---------------------------------------------------------------------------
__global__ __launch_bounds__(256) void cvt_x_kernel(const float* __restrict__ x,
                                                    _Float16* __restrict__ xt) {
  __shared__ float t[16][17];
  const int tx = threadIdx.x & 15;
  const int ty = threadIdx.x >> 4;
  const int tilesPerB = (CCH / 16) * (NPIX / 16);
  const int blk = blockIdx.x;
  const int b   = blk / tilesPerB;
  const int rem = blk % tilesPerB;
  const int c0  = (rem / (NPIX / 16)) * 16;
  const int n0  = (rem % (NPIX / 16)) * 16;
  t[ty][tx] = x[((size_t)b * CCH + c0 + ty) * NPIX + n0 + tx];
  __syncthreads();
  xt[((size_t)b * NPIX + n0 + ty) * CCH + c0 + tx] = (_Float16)t[tx][ty];
}

// Stage 1b: weights f32 -> f16 (Wq 32x256, Wk 32x256, Wv 256x256, row-major).
__global__ __launch_bounds__(256) void cvt_w_kernel(
    const float* __restrict__ Wq, const float* __restrict__ Wk,
    const float* __restrict__ Wv, _Float16* __restrict__ wqh,
    _Float16* __restrict__ wkh, _Float16* __restrict__ wvh) {
  const int nqk = DQKD * CCH;  // 8192
  int t = blockIdx.x * 256 + threadIdx.x;
  if (t < nqk) {
    wqh[t] = (_Float16)Wq[t];
  } else if (t < 2 * nqk) {
    wkh[t - nqk] = (_Float16)Wk[t - nqk];
  } else {
    int u = t - 2 * nqk;
    if (u < CCH * CCH) wvh[u] = (_Float16)Wv[u];
  }
}

// ---------------------------------------------------------------------------
// Stage 2: q^T, k^T = xt @ W^T  -> stored [B, N, 32] f16.
// A = xt tile (16 n x 32 c), B^T = W rows (16 d x 32 c). K-loop: C/32 = 8.
// Softmax scale 1/sqrt(32) folded into q here.
// One 16-row n-tile per wave, 4 waves/block.
// ---------------------------------------------------------------------------
__global__ __launch_bounds__(128) void proj_qk_kernel(
    const _Float16* __restrict__ xt, const _Float16* __restrict__ wqh,
    const _Float16* __restrict__ wkh, const float* __restrict__ bq,
    const float* __restrict__ bk, _Float16* __restrict__ qh,
    _Float16* __restrict__ kh) {
  const int wave = threadIdx.x >> 5;
  const int lane = threadIdx.x & 31;
  const int col  = lane & 15;
  const int hsel = lane >> 4;
  const int tile = blockIdx.x * 4 + wave;  // B*(N/16) = 1024 tiles
  const int b    = tile >> 8;
  const int n0   = (tile & 255) * 16;

  const _Float16* xrow = xt + ((size_t)b * NPIX + n0) * CCH;
  floatx8 aq0 = {}, aq1 = {}, ak0 = {}, ak1 = {};
#pragma unroll
  for (int kc = 0; kc < CCH; kc += 32) {
    halfx16 A   = load_tile16x32(xrow + kc, CCH);
    halfx16 Bq0 = load_tile16x32(wqh + kc, CCH);
    halfx16 Bq1 = load_tile16x32(wqh + 16 * CCH + kc, CCH);
    halfx16 Bk0 = load_tile16x32(wkh + kc, CCH);
    halfx16 Bk1 = load_tile16x32(wkh + 16 * CCH + kc, CCH);
    aq0 = wmma_fp16(A, Bq0, aq0);
    aq1 = wmma_fp16(A, Bq1, aq1);
    ak0 = wmma_fp16(A, Bk0, ak0);
    ak1 = wmma_fp16(A, Bk1, ak1);
  }
  const float scale = 0.17677669529663687f;  // 1/sqrt(DQK + 1e-8)
  const float bq0 = bq[col], bq1 = bq[16 + col];
  const float bk0 = bk[col], bk1 = bk[16 + col];
#pragma unroll
  for (int r = 0; r < 8; ++r) {
    const int n = n0 + r + 8 * hsel;  // D layout: row m = r + 8*(lane/16)
    _Float16* qp = qh + ((size_t)b * NPIX + n) * DQKD;
    _Float16* kp = kh + ((size_t)b * NPIX + n) * DQKD;
    qp[col]      = (_Float16)((aq0[r] + bq0) * scale);
    qp[16 + col] = (_Float16)((aq1[r] + bq1) * scale);
    kp[col]      = (_Float16)(ak0[r] + bk0);
    kp[16 + col] = (_Float16)(ak1[r] + bk1);
  }
}

// ---------------------------------------------------------------------------
// Stage 3: v = Wv @ x -> stored [B, C, N] f16 (exactly the B^T layout needed
// by the P@V^T WMMA in the attention kernel).
// A = Wv tile (16 cout x 32 c), B^T = xt tile (16 n x 32 c).
// One cout-tile x four n-tiles per wave (A-fragment reuse), 4 waves/block.
// ---------------------------------------------------------------------------
__global__ __launch_bounds__(128) void proj_v_kernel(
    const _Float16* __restrict__ xt, const _Float16* __restrict__ wvh,
    const float* __restrict__ bv, _Float16* __restrict__ vh) {
  const int wave = threadIdx.x >> 5;
  const int lane = threadIdx.x & 31;
  const int col  = lane & 15;
  const int hsel = lane >> 4;
  const int unit = blockIdx.x * 4 + wave;  // B*(C/16)*(N/64) = 4096 units
  const int b    = unit >> 10;
  const int rem  = unit & 1023;
  const int ct   = rem >> 6;
  const int n0   = (rem & 63) * 64;

  floatx8 acc[4];
#pragma unroll
  for (int t = 0; t < 4; ++t) acc[t] = (floatx8)0.0f;

#pragma unroll
  for (int kc = 0; kc < CCH; kc += 32) {
    halfx16 A = load_tile16x32(wvh + (size_t)(ct * 16) * CCH + kc, CCH);
#pragma unroll
    for (int t = 0; t < 4; ++t) {
      halfx16 Bt =
          load_tile16x32(xt + ((size_t)b * NPIX + n0 + t * 16) * CCH + kc, CCH);
      acc[t] = wmma_fp16(A, Bt, acc[t]);
    }
  }
#pragma unroll
  for (int t = 0; t < 4; ++t) {
#pragma unroll
    for (int r = 0; r < 8; ++r) {
      const int cout = ct * 16 + r + 8 * hsel;
      const int n    = n0 + t * 16 + col;
      vh[((size_t)b * CCH + cout) * NPIX + n] = (_Float16)(acc[t][r] + bv[cout]);
    }
  }
}

// ---------------------------------------------------------------------------
// Stage 4: flash attention + fused epilogue out = gamma * (P@V^T)/l + x.
// Per wave: TWO 16-row query tiles (32 q-rows) so every K/V fragment feeds
// 2x the WMMA work (L2 arithmetic intensity ~32 FLOP/byte). Per block: the
// 4 waves are kept in j-lockstep with __syncthreads() so their V-fragment
// loads of identical cachelines coalesce in the WGP cache instead of each
// hitting L2 (~4x L2 traffic cut). j processed 32 at a time:
//   4 WMMAs  S = Q K^T          (K = DQK = 32, single step, 2 q-tiles)
//   online softmax per tile: shfl-xor row reductions within 16-lane halves
//   P (D layout) -> A layout through per-wave LDS tiles, s_wait_dscnt fence
//   32 WMMAs O += P V^T         (16 channel tiles x 2 q-tiles)
// Accumulators: 2 x 16 x v8f = 256 VGPRs/lane, well under the 1024 budget.
// ---------------------------------------------------------------------------
__global__ __launch_bounds__(128) void attn_kernel(
    const _Float16* __restrict__ qh, const _Float16* __restrict__ kh,
    const _Float16* __restrict__ vh, const float* __restrict__ x,
    const float* __restrict__ gamma, float* __restrict__ out) {
  __shared__ _Float16 plds[4][2][16 * 32];
  const int wave  = threadIdx.x >> 5;
  const int lane  = threadIdx.x & 31;
  const int col   = lane & 15;
  const int hsel  = lane >> 4;
  const int tbase = blockIdx.x * 8 + wave * 2;  // B*(N/16) = 1024 tiles, 8/block
  const int b     = tbase >> 8;
  const int i0    = (tbase & 255) * 16;         // tiles i0 and i0+16

  const _Float16* qb = qh + ((size_t)b * NPIX + i0) * DQKD;
  const halfx16 Qf0 = load_tile16x32(qb, DQKD);
  const halfx16 Qf1 = load_tile16x32(qb + 16 * DQKD, DQKD);

  floatx8 acc0[16], acc1[16];
#pragma unroll
  for (int t = 0; t < 16; ++t) { acc0[t] = (floatx8)0.0f; acc1[t] = (floatx8)0.0f; }
  float m0[8], l0[8], m1[8], l1[8];
#pragma unroll
  for (int r = 0; r < 8; ++r) {
    m0[r] = -3.0e38f; l0[r] = 0.0f;
    m1[r] = -3.0e38f; l1[r] = 0.0f;
  }

  const _Float16* kb = kh + (size_t)b * NPIX * DQKD;
  const _Float16* vb = vh + (size_t)b * CCH * NPIX;
  _Float16* lds0 = &plds[wave][0][0];
  _Float16* lds1 = &plds[wave][1][0];
  const floatx8 zero = {};

  for (int j0 = 0; j0 < NPIX; j0 += 32) {
    // Keep the block's 4 waves on the same j-step for WGP$ V reuse.
    __syncthreads();

    halfx16 K0 = load_tile16x32(kb + (size_t)j0 * DQKD, DQKD);
    halfx16 K1 = load_tile16x32(kb + (size_t)(j0 + 16) * DQKD, DQKD);
    floatx8 Sa0 = wmma_fp16(Qf0, K0, zero);
    floatx8 Sa1 = wmma_fp16(Qf0, K1, zero);
    floatx8 Sb0 = wmma_fp16(Qf1, K0, zero);
    floatx8 Sb1 = wmma_fp16(Qf1, K1, zero);

    // Online softmax. D-layout: lane holds row i = r + 8*hsel, col = lane%16,
    // so each matrix row lives in one VGPR across a 16-lane half -> reduce
    // with shfl-xor 1/2/4/8 (stays inside the half on wave32).
    float pa0[8], pa1[8], alA[8], pb0[8], pb1[8], alB[8];
#pragma unroll
    for (int r = 0; r < 8; ++r) {
      float mx = fmaxf(Sa0[r], Sa1[r]);
      mx = fmaxf(mx, __shfl_xor(mx, 1, 32));
      mx = fmaxf(mx, __shfl_xor(mx, 2, 32));
      mx = fmaxf(mx, __shfl_xor(mx, 4, 32));
      mx = fmaxf(mx, __shfl_xor(mx, 8, 32));
      const float mn = fmaxf(m0[r], mx);
      alA[r] = __expf(m0[r] - mn);
      m0[r]  = mn;
      pa0[r] = __expf(Sa0[r] - mn);
      pa1[r] = __expf(Sa1[r] - mn);
      float rs = pa0[r] + pa1[r];
      rs += __shfl_xor(rs, 1, 32);
      rs += __shfl_xor(rs, 2, 32);
      rs += __shfl_xor(rs, 4, 32);
      rs += __shfl_xor(rs, 8, 32);
      l0[r] = l0[r] * alA[r] + rs;
    }
#pragma unroll
    for (int r = 0; r < 8; ++r) {
      float mx = fmaxf(Sb0[r], Sb1[r]);
      mx = fmaxf(mx, __shfl_xor(mx, 1, 32));
      mx = fmaxf(mx, __shfl_xor(mx, 2, 32));
      mx = fmaxf(mx, __shfl_xor(mx, 4, 32));
      mx = fmaxf(mx, __shfl_xor(mx, 8, 32));
      const float mn = fmaxf(m1[r], mx);
      alB[r] = __expf(m1[r] - mn);
      m1[r]  = mn;
      pb0[r] = __expf(Sb0[r] - mn);
      pb1[r] = __expf(Sb1[r] - mn);
      float rs = pb0[r] + pb1[r];
      rs += __shfl_xor(rs, 1, 32);
      rs += __shfl_xor(rs, 2, 32);
      rs += __shfl_xor(rs, 4, 32);
      rs += __shfl_xor(rs, 8, 32);
      l1[r] = l1[r] * alB[r] + rs;
    }
#pragma unroll
    for (int t = 0; t < 16; ++t)
#pragma unroll
      for (int r = 0; r < 8; ++r) {
        acc0[t][r] *= alA[r];
        acc1[t][r] *= alB[r];
      }

    // P: D layout -> A-fragment layout via per-wave LDS bounce. DS ops from
    // one wave execute in order; the explicit dscnt wait + memory clobbers
    // fence the cross-lane dependence the compiler cannot see.
#pragma unroll
    for (int r = 0; r < 8; ++r) {
      const int row = r + 8 * hsel;
      lds0[row * 32 + col]      = (_Float16)pa0[r];
      lds0[row * 32 + 16 + col] = (_Float16)pa1[r];
      lds1[row * 32 + col]      = (_Float16)pb0[r];
      lds1[row * 32 + 16 + col] = (_Float16)pb1[r];
    }
    asm volatile("s_wait_dscnt 0x0" ::: "memory");
    halfx16 Pf0 = load_tile16x32(lds0, 32);
    halfx16 Pf1 = load_tile16x32(lds1, 32);
    asm volatile("" ::: "memory");

#pragma unroll
    for (int t = 0; t < 16; ++t) {
      halfx16 Vf = load_tile16x32(vb + (size_t)(t * 16) * NPIX + j0, NPIX);
      acc0[t] = wmma_fp16(Pf0, Vf, acc0[t]);
      acc1[t] = wmma_fp16(Pf1, Vf, acc1[t]);
    }
  }

  const float g = gamma[0];
  float il0[8], il1[8];
#pragma unroll
  for (int r = 0; r < 8; ++r) { il0[r] = 1.0f / l0[r]; il1[r] = 1.0f / l1[r]; }
#pragma unroll
  for (int t = 0; t < 16; ++t) {
    const int c = t * 16 + col;
    const size_t base = ((size_t)b * CCH + c) * NPIX + i0 + 8 * hsel;
    const float* xp0 = x + base;
    float* op0 = out + base;
#pragma unroll
    for (int r = 0; r < 8; ++r) op0[r] = g * acc0[t][r] * il0[r] + xp0[r];
    const float* xp1 = x + base + 16;
    float* op1 = out + base + 16;
#pragma unroll
    for (int r = 0; r < 8; ++r) op1[r] = g * acc1[t][r] * il1[r] + xp1[r];
  }
}

// ---------------------------------------------------------------------------
extern "C" void kernel_launch(void* const* d_in, const int* in_sizes, int n_in,
                              void* d_out, int out_size, void* d_ws,
                              size_t ws_size, hipStream_t stream) {
  const float* x     = (const float*)d_in[0];
  const float* Wq    = (const float*)d_in[1];
  const float* bq    = (const float*)d_in[2];
  const float* Wk    = (const float*)d_in[3];
  const float* bk    = (const float*)d_in[4];
  const float* Wv    = (const float*)d_in[5];
  const float* bv    = (const float*)d_in[6];
  const float* gamma = (const float*)d_in[7];
  float* out = (float*)d_out;

  // Workspace carve-up (~19.0 MB total).
  char* ws = (char*)d_ws;
  size_t off = 0;
  _Float16* xt  = (_Float16*)(ws + off); off += (size_t)BATCH * NPIX * CCH * 2;
  _Float16* wqh = (_Float16*)(ws + off); off += (size_t)DQKD * CCH * 2;
  _Float16* wkh = (_Float16*)(ws + off); off += (size_t)DQKD * CCH * 2;
  _Float16* wvh = (_Float16*)(ws + off); off += (size_t)CCH * CCH * 2;
  _Float16* qh  = (_Float16*)(ws + off); off += (size_t)BATCH * NPIX * DQKD * 2;
  _Float16* kh  = (_Float16*)(ws + off); off += (size_t)BATCH * NPIX * DQKD * 2;
  _Float16* vh  = (_Float16*)(ws + off); off += (size_t)BATCH * CCH * NPIX * 2;
  (void)off; (void)ws_size; (void)in_sizes; (void)n_in; (void)out_size;

  cvt_x_kernel<<<BATCH * (CCH / 16) * (NPIX / 16), 256, 0, stream>>>(x, xt);
  cvt_w_kernel<<<(2 * DQKD * CCH + CCH * CCH + 255) / 256, 256, 0, stream>>>(
      Wq, Wk, Wv, wqh, wkh, wvh);
  proj_qk_kernel<<<BATCH * (NPIX / 16) / 4, 128, 0, stream>>>(xt, wqh, wkh, bq,
                                                              bk, qh, kh);
  proj_v_kernel<<<BATCH * (CCH / 16) * (NPIX / 64) / 4, 128, 0, stream>>>(
      xt, wvh, bv, vh);
  // 1024 query tiles / (4 waves x 2 tiles) = 128 blocks.
  attn_kernel<<<BATCH * (NPIX / 16) / 8, 128, 0, stream>>>(qh, kh, vh, x,
                                                           gamma, out);
}
